// LOIMLoss_40690520162428
// MI455X (gfx1250) — matrix operand order repaired and checked by reading, not compiled, not run on hardware
//
#include <hip/hip_runtime.h>
#include <hip/hip_bf16.h>

typedef __attribute__((ext_vector_type(8)))  _Float16 v8h;
typedef __attribute__((ext_vector_type(16))) _Float16 v16h;
typedef __attribute__((ext_vector_type(8)))  float    v8f;

#define NF        256      // features (K)
#define NPIDS     5532
#define NCQ       5000
#define NCLS      (NPIDS + NCQ)        // 10532
#define NTILES    ((NCLS + 15) / 16)   // 659
#define CPAD      (NTILES * 16)        // 10544
#define NROWS     8192
#define OIM_SCALE 30.0f
#define NEG_BIG   (-1.0e30f)

// ---------------------------------------------------------------- utilities
__device__ __forceinline__ void split_f16(float x, _Float16& h, _Float16& l) {
    _Float16 hh = (_Float16)x;
    h = hh;
    l = (_Float16)(x - (float)hh);
}

// ---------------------------------------------------------------- kernel 0
__global__ void oim_zero_out(float* out) {
    if (threadIdx.x == 0 && blockIdx.x == 0) out[0] = 0.0f;
}

// ---------------------------------------------------------------- kernel 1
// Split inputs (NROWS x NF f32, row major) into f16 hi/lo, same layout.
__global__ void oim_split_inputs(const float* __restrict__ x,
                                 _Float16* __restrict__ a_hi,
                                 _Float16* __restrict__ a_lo) {
    int i = blockIdx.x * blockDim.x + threadIdx.x;       // NROWS*NF threads
    float v = x[i];
    split_f16(v, a_hi[i], a_lo[i]);
}

// ---------------------------------------------------------------- kernel 2
// Split + pad + pre-swizzle W = [lut; cq] into B-fragment layout:
//   wt[tile*4096 + ks*512 + lane*16 + s] = W[tile*16 + (lane&15)][ks*32 + k(lane>>4, s)]
// with k(h,s) = 8h+s (s<8), 16+8h+s-8 (s>=8); each lane's fragment is 16 contiguous halves.
__global__ void oim_split_weights(const float* __restrict__ lut,
                                  const float* __restrict__ cq,
                                  _Float16* __restrict__ wt_hi,
                                  _Float16* __restrict__ wt_lo) {
    int i = blockIdx.x * blockDim.x + threadIdx.x;       // NTILES*NF*16 threads
    int j = (i >> 4) & 15;       // column within tile (class)
    int k = ((i >> 9) & 7) * 32 + ((i >> 8) & 1) * 8 + (i & 7) + ((i >> 3) & 1) * 16; // feature
    int c = i >> 12;             // tile  (16*256 = 4096 = 2^12)
    int col = c * 16 + j;
    float v = 0.0f;
    if (col < NPIDS)       v = lut[(size_t)col * NF + k];
    else if (col < NCLS)   v = cq[(size_t)(col - NPIDS) * NF + k];
    split_f16(v, wt_hi[i], wt_lo[i]);
}

// ---------------------------------------------------------------- kernel 3
// One block = 16 rows, 4 wave32s interleave over the 659 column tiles.
// Online softmax carried in registers, combined across lanes/waves at the end.
__global__ void __launch_bounds__(128)
oim_loss_main(const _Float16* __restrict__ a_hi,
              const _Float16* __restrict__ a_lo,
              const _Float16* __restrict__ wt_hi,
              const _Float16* __restrict__ wt_lo,
              const int* __restrict__ targets,
              float* __restrict__ out) {
    __shared__ int   lds_label[16];
    __shared__ float lds_t[16];          // scaled logit at label (unique writer/row)
    __shared__ float lds_m[4][16];
    __shared__ float lds_s[4][16];

    const int tid     = threadIdx.x;
    const int lane    = tid & 31;
    const int wave    = tid >> 5;        // 0..3
    const int grp     = lane >> 4;       // 0: rows 0-7, 1: rows 8-15
    const int n       = lane & 15;       // output column within tile
    const int rowbase = blockIdx.x * 16;

    if (tid < 16) lds_label[tid] = targets[rowbase + tid] - 1;
    __syncthreads();

    // ---- load A fragments for all 8 k-steps (hi and lo) into registers ----
    // 16-bit A 16x32 layout: lane holds M=lane&15; halves 0..7 = K=k0+grp*8..,
    // halves 8..15 = K=k0+16+grp*8..  (grp = lane>>4)
    const size_t arow = (size_t)(rowbase + (lane & 15)) * NF;
    v16h ah[8], al[8];
#pragma unroll
    for (int ks = 0; ks < 8; ++ks) {
        const int k0 = ks * 32;
        v8h h0 = *(const v8h*)(a_hi + arow + k0 + grp * 8);
        v8h h1 = *(const v8h*)(a_hi + arow + k0 + 16 + grp * 8);
        v8h l0 = *(const v8h*)(a_lo + arow + k0 + grp * 8);
        v8h l1 = *(const v8h*)(a_lo + arow + k0 + 16 + grp * 8);
#pragma unroll
        for (int i = 0; i < 8; ++i) {
            ah[ks][i] = h0[i]; ah[ks][8 + i] = h1[i];
            al[ks][i] = l0[i]; al[ks][8 + i] = l1[i];
        }
    }

    // ---- online softmax state: 8 rows per lane (rows r + 8*grp) ----
    float mr[8], sr[8];
#pragma unroll
    for (int r = 0; r < 8; ++r) { mr[r] = NEG_BIG; sr[r] = 0.0f; }

    // ---- sweep column tiles ----
    for (int c = wave; c < NTILES; c += 4) {
        const size_t btile = (size_t)c * NF * 16 + (size_t)lane * 16; // + ks*32*16
        v8f acc = {};
#pragma unroll
        for (int ks = 0; ks < 8; ++ks) {
            v16h bh = *(const v16h*)(wt_hi + btile + (size_t)ks * 512);
            v16h bl = *(const v16h*)(wt_lo + btile + (size_t)ks * 512);
            // split-f16: hi*hi + lo*hi + hi*lo  (~f32 accuracy)
            acc = __builtin_amdgcn_wmma_f32_16x16x32_f16(
                      false, ah[ks], false, bh, (short)0, acc, false, false);
            acc = __builtin_amdgcn_wmma_f32_16x16x32_f16(
                      false, al[ks], false, bh, (short)0, acc, false, false);
            acc = __builtin_amdgcn_wmma_f32_16x16x32_f16(
                      false, ah[ks], false, bl, (short)0, acc, false, false);
        }

        const int col = c * 16 + n;
        if (col < NCLS) {
#pragma unroll
            for (int r = 0; r < 8; ++r) {
                float v = acc[r] * OIM_SCALE;
                int m = r + 8 * grp;                 // row within the 16-row tile
                if (lds_label[m] == col) lds_t[m] = v;   // unique writer per row
                float mn = fmaxf(mr[r], v);
                sr[r] = sr[r] * __expf(mr[r] - mn) + __expf(v - mn);
                mr[r] = mn;
            }
        }
    }

    // ---- reduce (m,s) across the 16 lanes that share each row ----
#pragma unroll
    for (int r = 0; r < 8; ++r) {
#pragma unroll
        for (int off = 1; off < 16; off <<= 1) {
            float om = __shfl_xor(mr[r], off, 32);
            float os = __shfl_xor(sr[r], off, 32);
            float mn = fmaxf(mr[r], om);
            sr[r] = sr[r] * __expf(mr[r] - mn) + os * __expf(om - mn);
            mr[r] = mn;
        }
    }
    if ((lane & 15) == 0) {           // lane 0 -> rows 0-7, lane 16 -> rows 8-15
#pragma unroll
        for (int r = 0; r < 8; ++r) {
            int m = r + 8 * grp;
            lds_m[wave][m] = mr[r];
            lds_s[wave][m] = sr[r];
        }
    }
    __syncthreads();

    // ---- combine the 4 wave partials, compute NLL, accumulate loss ----
    if (tid < 16) {
        int row = tid;
        float M = NEG_BIG, S = 0.0f;
#pragma unroll
        for (int w = 0; w < 4; ++w) {
            float om = lds_m[w][row], os = lds_s[w][row];
            float mn = fmaxf(M, om);
            S = S * __expf(M - mn) + os * __expf(om - mn);
            M = mn;
        }
        float nll = M + __logf(S) - lds_t[row];
#pragma unroll
        for (int off = 1; off < 16; off <<= 1)
            nll += __shfl_xor(nll, off, 32);
        if (tid == 0)
            atomicAdd(out, nll * (1.0f / (float)NROWS));
    }
}

// ---------------------------------------------------------------- launcher
extern "C" void kernel_launch(void* const* d_in, const int* in_sizes, int n_in,
                              void* d_out, int out_size, void* d_ws, size_t ws_size,
                              hipStream_t stream) {
    (void)in_sizes; (void)n_in; (void)out_size; (void)ws_size;
    const float* x       = (const float*)d_in[0];
    const int*   targets = (const int*)d_in[1];
    // d_in[2] = ious (unused by the reference loss)
    const float* lut     = (const float*)d_in[3];
    const float* cq      = (const float*)d_in[4];
    float*       out     = (float*)d_out;

    _Float16* a_hi  = (_Float16*)d_ws;                      // NROWS*NF
    _Float16* a_lo  = a_hi  + (size_t)NROWS * NF;           // NROWS*NF
    _Float16* wt_hi = a_lo  + (size_t)NROWS * NF;           // CPAD*NF
    _Float16* wt_lo = wt_hi + (size_t)CPAD * NF;            // CPAD*NF

    oim_zero_out<<<1, 32, 0, stream>>>(out);

    {   // split inputs: NROWS*NF elements, 1 per thread
        int total = NROWS * NF;
        oim_split_inputs<<<total / 256, 256, 0, stream>>>(x, a_hi, a_lo);
    }
    {   // split + transpose weights: NTILES*NF*16 elements
        int total = NTILES * NF * 16;
        oim_split_weights<<<total / 256, 256, 0, stream>>>(lut, cq, wt_hi, wt_lo);
    }
    oim_loss_main<<<NROWS / 16, 128, 0, stream>>>(a_hi, a_lo, wt_hi, wt_lo,
                                                  targets, out);
}
